// pLayer_39599598469767
// MI455X (gfx1250) — matrix operand order repaired and checked by reading, not compiled, not run on hardware
//
#include <hip/hip_runtime.h>

// ---------------- problem constants (match reference) ----------------
#define N_INX   2048
#define N_OUTX  2048
#define BATCHX  4096
#define M_EXTX  2050            // N_IN + 2
#define GMIN_C  0.01f
#define GMAX_C  0.1f
#define PGMIN_C 1e-4f
#define ETA2_C  0.1f

typedef __attribute__((ext_vector_type(16))) __bf16 v16bf;
typedef __attribute__((ext_vector_type(8)))  __bf16 v8bf;
typedef __attribute__((ext_vector_type(8)))  float  v8f;

// ---------------- small helpers ----------------
__device__ __forceinline__ float clip_zero(float t) {
    float th = fminf(fmaxf(t, -GMAX_C), GMAX_C);
    return (fabsf(th) < GMIN_C) ? 0.0f : th;
}

// A-matrix 16x32 bf16 fragment (ISA 7.12.2): per lane, row M = lane%16,
// K = kb..kb+7 (chunk0) and kb+16..kb+23 (chunk1), kb = (lane<16?0:8).
// Caller passes pointer already at (row*K + k + kb); chunks at +0 and +16.
__device__ __forceinline__ v16bf load_frag_a(const __bf16* p) {
    v8bf c0 = *(const v8bf*)(p);
    v8bf c1 = *(const v8bf*)(p + 16);
    v16bf r;
#pragma unroll
    for (int i = 0; i < 8; ++i) { r[i] = c0[i]; r[i + 8] = c1[i]; }
    return r;
}

// B-matrix 32x16 bf16 fragment: per lane, col N = lane%16,
// K = kb..kb+7 (chunk0) and kb+16..kb+23 (chunk1), kb = (lane<16?0:8).
// We store B pre-transposed (BT[n][k]); chunks at +0 and +16.
__device__ __forceinline__ v16bf load_frag_b(const __bf16* p) {
    v8bf c0 = *(const v8bf*)(p);
    v8bf c1 = *(const v8bf*)(p + 16);
    v16bf r;
#pragma unroll
    for (int i = 0; i < 8; ++i) { r[i] = c0[i]; r[i + 8] = c1[i]; }
    return r;
}

// ---------------- kernel 1: column sums of a^2 ----------------
__global__ void k_sx2(const float* __restrict__ a, float* __restrict__ sx2) {
    int m = blockIdx.x * blockDim.x + threadIdx.x;
    if (m < N_INX) {
        float s = 0.f;
        for (int e = 0; e < BATCHX; ++e) {
            float v = a[(size_t)e * N_INX + m];
            s = fmaf(v, v, s);
        }
        sx2[m] = s;
    }
    if (m == 0) { sx2[N_INX] = (float)BATCHX; sx2[N_INX + 1] = 0.f; }
}

// ---------------- kernel 2: bf16 hi/lo split of a ----------------
__global__ void k_convert_a(const float* __restrict__ a,
                            __bf16* __restrict__ ahi, __bf16* __restrict__ alo) {
    size_t stride = (size_t)gridDim.x * blockDim.x;
    size_t n = (size_t)BATCHX * N_INX;
    for (size_t i = (size_t)blockIdx.x * blockDim.x + threadIdx.x; i < n; i += stride) {
        float v = a[i];
        __bf16 h = (__bf16)v;
        ahi[i] = h;
        alo[i] = (__bf16)(v - (float)h);
    }
}

// ---------------- kernel 3: per-column theta preprocessing ----------------
// thread n: gmin, colsum(|th_eff|), writes transposed W^T and G'^T (hi/lo),
// bias[n], g2048[n], g_col[n], and term-1 partial t1col[n].
__global__ void k_prep_theta(const float* __restrict__ theta, const float* __restrict__ sx2,
                             __bf16* __restrict__ wt_hi, __bf16* __restrict__ wt_lo,
                             __bf16* __restrict__ gt_hi, __bf16* __restrict__ gt_lo,
                             float* __restrict__ bias,  float* __restrict__ g2048,
                             float* __restrict__ gcol,  float* __restrict__ t1col) {
    int n = blockIdx.x * blockDim.x + threadIdx.x;
    if (n >= N_OUTX) return;
    float gmin = 3.402823466e38f, colsum = 0.f, colabs = 0.f, colacc = 0.f;
    for (int m = 0; m < M_EXTX; ++m) {
        float t  = theta[(size_t)m * N_OUTX + n];
        float at = fabsf(t);
        gmin   = fminf(gmin, at);
        colsum += fabsf(clip_zero(t));
        colabs += at;
        colacc  = fmaf(sx2[m], at, colacc);
    }
    float invs   = (colsum > 0.f) ? (1.f / colsum) : 0.f;
    float gscale = PGMIN_C / gmin;          // same div-by-zero semantics as ref
    gcol[n]  = gscale * colabs;
    t1col[n] = gscale * colacc;

    for (int m = 0; m < N_INX; ++m) {
        float t  = theta[(size_t)m * N_OUTX + n];
        float th = clip_zero(t);
        size_t idx = (size_t)n * N_INX + m;           // transposed layout
        float w  = th * invs;
        __bf16 wh = (__bf16)w;
        wt_hi[idx] = wh; wt_lo[idx] = (__bf16)(w - (float)wh);
        float sgn = (th >= 0.f) ? 1.f : -1.f;
        float gp  = gscale * fabsf(t) * sgn;
        __bf16 gh = (__bf16)gp;
        gt_hi[idx] = gh; gt_lo[idx] = (__bf16)(gp - (float)gh);
    }
    {   // virtual ones-row m == N_IN
        float t  = theta[(size_t)N_INX * N_OUTX + n];
        float th = clip_zero(t);
        bias[n]  = th * invs;
        float sgn = (th >= 0.f) ? 1.f : -1.f;
        g2048[n] = gscale * fabsf(t) * sgn;
    }
}

// ---------------- kernel 4: bf16x3 WMMA GEMM ----------------
// D[4096,2048] = A[4096,2048] @ B[2048,2048] with A row-major hi/lo and
// B transposed (BT[n][k]) hi/lo.  mode 0: Z = D + bias.  mode 1: atomicAdd
// of sum(D .* Z) into t2_accum (Cx-free power cross term).
__global__ __launch_bounds__(256)
void k_gemm_bf16x3(const __bf16* __restrict__ Ahi, const __bf16* __restrict__ Alo,
                   const __bf16* __restrict__ BThi, const __bf16* __restrict__ BTlo,
                   const float* __restrict__ bias, float* __restrict__ Z,
                   float* __restrict__ t2_accum, int mode) {
    const int K = N_INX;
    const int lane = threadIdx.x & 31;
    const int wave = threadIdx.x >> 5;
    const int wM = wave >> 2, wN = wave & 3;          // 2 x 4 wave grid
    const int tileM = blockIdx.y * 128 + wM * 64;     // 64 rows  per wave
    const int tileN = blockIdx.x * 128 + wN * 32;     // 32 cols  per wave
    const int lr    = lane & 15;
    const int aKoff = (lane >> 4) * 8;
    const int bKoff = (lane >> 4) * 8;

    v8f c[4][2] = {};

    for (int k = 0; k < K; k += 32) {
        v16bf ah[4], al[4], bh[2], bl[2];
#pragma unroll
        for (int i = 0; i < 4; ++i) {
            size_t off = (size_t)(tileM + i * 16 + lr) * K + k + aKoff;
            ah[i] = load_frag_a(Ahi + off);
            al[i] = load_frag_a(Alo + off);
            // unconditional speculative prefetch of next K-slab (arrays are
            // padded)
            __builtin_prefetch(Ahi + off + 32, 0, 3);
            __builtin_prefetch(Alo + off + 32, 0, 3);
        }
#pragma unroll
        for (int j = 0; j < 2; ++j) {
            size_t off = (size_t)(tileN + j * 16 + lr) * K + k + bKoff;
            bh[j] = load_frag_b(BThi + off);
            bl[j] = load_frag_b(BTlo + off);
            __builtin_prefetch(BThi + off + 32, 0, 3);
            __builtin_prefetch(BTlo + off + 32, 0, 3);
        }
#pragma unroll
        for (int i = 0; i < 4; ++i)
#pragma unroll
            for (int j = 0; j < 2; ++j) {
                // fp32 emulation: hi*hi + hi*lo + lo*hi, f32 accumulate
                c[i][j] = __builtin_amdgcn_wmma_f32_16x16x32_bf16(
                    false, ah[i], false, bh[j], (short)0, c[i][j], false, false);
                c[i][j] = __builtin_amdgcn_wmma_f32_16x16x32_bf16(
                    false, ah[i], false, bl[j], (short)0, c[i][j], false, false);
                c[i][j] = __builtin_amdgcn_wmma_f32_16x16x32_bf16(
                    false, al[i], false, bh[j], (short)0, c[i][j], false, false);
            }
    }

    const int rowOff = (lane >> 4) * 8;   // C/D layout: VGPR r -> M = r (+8 for hi lanes)
    if (mode == 0) {
#pragma unroll
        for (int i = 0; i < 4; ++i)
#pragma unroll
            for (int j = 0; j < 2; ++j) {
                int n = tileN + j * 16 + lr;
                float b = bias[n];
#pragma unroll
                for (int r = 0; r < 8; ++r) {
                    int m = tileM + i * 16 + rowOff + r;
                    Z[(size_t)m * N_OUTX + n] = c[i][j][r] + b;
                }
            }
    } else {
        float part = 0.f;
#pragma unroll
        for (int i = 0; i < 4; ++i)
#pragma unroll
            for (int j = 0; j < 2; ++j) {
                int n = tileN + j * 16 + lr;
#pragma unroll
                for (int r = 0; r < 8; ++r) {
                    int e = tileM + i * 16 + rowOff + r;
                    part = fmaf(c[i][j][r], Z[(size_t)e * N_OUTX + n], part);
                }
            }
        __shared__ float red[256];
        red[threadIdx.x] = part;
        __syncthreads();
        for (int s = 128; s > 0; s >>= 1) {
            if ((int)threadIdx.x < s) red[threadIdx.x] += red[threadIdx.x + s];
            __syncthreads();
        }
        if (threadIdx.x == 0) atomicAdd(t2_accum, red[0]);
    }
}

// ---------------- kernel 5: per-column reductions of z ----------------
__global__ void k_zreduce(const float* __restrict__ Z, float* __restrict__ Y2,
                          float* __restrict__ zsum, float* __restrict__ t2) {
    int n = blockIdx.x * blockDim.x + threadIdx.x;
    if (n < N_OUTX) {
        float s = 0.f, s2 = 0.f;
        for (int e = 0; e < BATCHX; ++e) {
            float v = Z[(size_t)e * N_OUTX + n];
            s += v;
            s2 = fmaf(v, v, s2);
        }
        zsum[n] = s; Y2[n] = s2;
    }
    if (blockIdx.x == 0 && threadIdx.x == 0) *t2 = 0.f;   // zero before gemm2
}

// ---------------- kernel 6: assemble power ----------------
__global__ void k_finalize(const float* __restrict__ t1col, const float* __restrict__ g2048,
                           const float* __restrict__ zsum,  const float* __restrict__ Y2,
                           const float* __restrict__ gcol,  const float* __restrict__ t2,
                           float* __restrict__ out_power) {
    __shared__ float red[256];
    float acc = 0.f;
    for (int n = threadIdx.x; n < N_OUTX; n += 256)
        acc += t1col[n] + Y2[n] * gcol[n] - 2.f * g2048[n] * zsum[n];
    red[threadIdx.x] = acc;
    __syncthreads();
    for (int s = 128; s > 0; s >>= 1) {
        if ((int)threadIdx.x < s) red[threadIdx.x] += red[threadIdx.x + s];
        __syncthreads();
    }
    if (threadIdx.x == 0) out_power[0] = (red[0] - 2.f * t2[0]) / (float)BATCHX;
}

// ---------------- host launcher ----------------
extern "C" void kernel_launch(void* const* d_in, const int* in_sizes, int n_in,
                              void* d_out, int out_size, void* d_ws, size_t ws_size,
                              hipStream_t stream) {
    (void)in_sizes; (void)n_in; (void)out_size; (void)ws_size;
    const float* a     = (const float*)d_in[0];   // [BATCH, N_IN]
    const float* theta = (const float*)d_in[1];   // [N_IN+2, N_OUT]
    float* z     = (float*)d_out;                       // [BATCH * N_OUT]
    float* power = z + (size_t)BATCHX * N_OUTX;         // 1 element

    char* ws = (char*)d_ws;
    auto take = [&](size_t bytes) -> char* {
        char* p = ws;
        ws += (bytes + 255 + 256) & ~(size_t)255;  // +256B pad: prefetch overrun safety
        return p;
    };
    __bf16* ahi  = (__bf16*)take((size_t)BATCHX * N_INX * 2);
    __bf16* alo  = (__bf16*)take((size_t)BATCHX * N_INX * 2);
    __bf16* wthi = (__bf16*)take((size_t)N_OUTX * N_INX * 2);
    __bf16* wtlo = (__bf16*)take((size_t)N_OUTX * N_INX * 2);
    __bf16* gthi = (__bf16*)take((size_t)N_OUTX * N_INX * 2);
    __bf16* gtlo = (__bf16*)take((size_t)N_OUTX * N_INX * 2);
    float* sx2   = (float*)take(M_EXTX * 4);
    float* bias  = (float*)take(N_OUTX * 4);
    float* g2048 = (float*)take(N_OUTX * 4);
    float* gcol  = (float*)take(N_OUTX * 4);
    float* t1col = (float*)take(N_OUTX * 4);
    float* Y2    = (float*)take(N_OUTX * 4);
    float* zsum  = (float*)take(N_OUTX * 4);
    float* t2    = (float*)take(256);

    k_sx2      <<<N_INX / 256, 256, 0, stream>>>(a, sx2);
    k_convert_a<<<2048, 256, 0, stream>>>(a, ahi, alo);
    k_prep_theta<<<N_OUTX / 256, 256, 0, stream>>>(theta, sx2, wthi, wtlo, gthi, gtlo,
                                                   bias, g2048, gcol, t1col);
    dim3 ggrid(N_OUTX / 128, BATCHX / 128);  // (16, 32)
    k_gemm_bf16x3<<<ggrid, 256, 0, stream>>>(ahi, alo, wthi, wtlo, bias, z, t2, 0);
    k_zreduce  <<<N_OUTX / 256, 256, 0, stream>>>(z, Y2, zsum, t2);
    k_gemm_bf16x3<<<ggrid, 256, 0, stream>>>(ahi, alo, gthi, gtlo, bias, z, t2, 1);
    k_finalize <<<1, 256, 0, stream>>>(t1col, g2048, zsum, Y2, gcol, t2, power);
}
